// VectorQuantize_89077621719083
// MI455X (gfx1250) — compile-verified, matrix operand order checked by reading, not run on hardware
//
#include <hip/hip_runtime.h>
#include <hip/hip_bf16.h>

// ---------------- problem constants ----------------
#define Bq   8
#define Nq   4096
#define DIMq 1024
#define CDq  512
#define Kq   8192
#define Mtot (Bq * Nq)   // 32768 rows

typedef __bf16 bf16;
typedef __attribute__((ext_vector_type(16))) __bf16 v16bf;
typedef __attribute__((ext_vector_type(8)))  __bf16 v8bf;
typedef __attribute__((ext_vector_type(8)))  float  v8f;
typedef __attribute__((ext_vector_type(4)))  float  v4f;
typedef __attribute__((ext_vector_type(4)))  unsigned int v4u;
typedef __attribute__((ext_vector_type(8)))  int    v8i;
typedef __attribute__((ext_vector_type(4)))  int    v4i;

// address-space-qualified vector pointers for the async-LDS builtin
typedef __attribute__((address_space(1))) v4i av4i_g;   // global
typedef __attribute__((address_space(3))) v4i av4i_l;   // LDS

// ---------------- workspace layout (bytes) ----------------
static const size_t WS_WINT  = 0;                      // [512 x 1024] bf16 (W_in^T)   1 MB
static const size_t WS_WOUTT = (size_t)1  << 20;       // [1024 x 512] bf16 (W_out^T)  1 MB
static const size_t WS_EMBED = (size_t)2  << 20;       // [8192 x 512] bf16            8 MB
static const size_t WS_HALF  = (size_t)10 << 20;       // [8192] f32  (0.5*||e||^2)
static const size_t WS_IND   = ((size_t)10 << 20) + (1u << 16); // [32768] i32
static const size_t WS_XPB   = (size_t)11 << 20;       // [32768 x 512] bf16 (xp)     32 MB
static const size_t WS_PART  = (size_t)44 << 20;       // [32768] f32 loss partials

#define LDS_BYTE_OFF(p) ((unsigned)(size_t)(__attribute__((address_space(3))) char*)(void*)(p))

// Build a 16-element bf16 WMMA fragment from two contiguous 16B chunks:
// elements [0..7] = p[0..7] (K = base..base+7), [8..15] = p[16..23] (K = base+16..base+23).
__device__ __forceinline__ v16bf frag16(const bf16* p) {
    v8bf lo = *(const v8bf*)p;
    v8bf hi = *(const v8bf*)(p + 16);
    v16bf f;
#pragma unroll
    for (int i = 0; i < 8; ++i) { f[i] = lo[i]; f[8 + i] = hi[i]; }
    return f;
}

__device__ __forceinline__ v8f wmma_bf16(v16bf a, v16bf b, v8f c) {
    return __builtin_amdgcn_wmma_f32_16x16x32_bf16(false, a, false, b, (short)0, c, false, false);
}

// ---------------- async global->LDS staging (ASYNCcnt path) ----------------
#if __has_builtin(__builtin_amdgcn_global_load_async_to_lds_b128)
#define HAS_ASYNC 1
__device__ __forceinline__ void stage_b128(bf16* ldst, const bf16* gsrc) {
    __builtin_amdgcn_global_load_async_to_lds_b128(
        (av4i_g*)(uintptr_t)gsrc,
        (av4i_l*)(__attribute__((address_space(3))) char*)(void*)ldst,
        0, 0);
}
__device__ __forceinline__ void stage_wait() {
    asm volatile("s_wait_asynccnt 0x0" ::: "memory");
}
#else
#define HAS_ASYNC 0
__device__ __forceinline__ void stage_b128(bf16* ldst, const bf16* gsrc) {
    *(v8bf*)ldst = *(const v8bf*)gsrc;
}
__device__ __forceinline__ void stage_wait() {}
#endif

// ---------------- Tensor Data Mover: 2-D bf16 tile -> LDS ----------------
// Loads [rows x 512] bf16 tile (global row stride 512 elems) into LDS with a
// 520-element (1040 B) destination row stride produced by the TDM pad engine:
// pad_interval = 256 dwords (one 1024 B row), pad_amount = 4 dwords (16 B).
#if __has_builtin(__builtin_amdgcn_tensor_load_to_lds)
#define HAS_TDM 1
__device__ __forceinline__ void tdm_load_2d(unsigned lds_byte_off, const bf16* gsrc, int rows) {
    unsigned long long ga = (unsigned long long)(uintptr_t)gsrc;
    v4u g0;
    g0[0] = 1u;                                           // count=1, user descriptor
    g0[1] = lds_byte_off;                                 // lds_addr
    g0[2] = (unsigned)(ga & 0xFFFFFFFFull);               // global_addr lo
    g0[3] = (unsigned)((ga >> 32) & 0x01FFFFFFull) | 0x80000000u; // addr hi | type=2
    v8i g1;
    g1[0] = (1 << 16) | (1 << 20) | (7 << 22) | (3 << 25); // data_size=2B, pad_en, 256dw/4dw
    g1[1] = (int)(512u << 16);                            // tensor_dim0 = 512
    g1[2] = (int)((unsigned)rows << 16);                  // tensor_dim1 = rows
    g1[3] = (int)(512u << 16);                            // tile_dim0 = 512
    g1[4] = rows;                                         // tile_dim1 = rows, tile_dim2=0
    g1[5] = 512;                                          // tensor_dim0_stride lo
    g1[6] = 0;
    g1[7] = 0;
    v4i gz = {0, 0, 0, 0};
#if defined(__clang_major__) && (__clang_major__ >= 23)
    v8i gz8 = {0, 0, 0, 0, 0, 0, 0, 0};
    __builtin_amdgcn_tensor_load_to_lds(g0, g1, gz, gz, gz8, 0);
#else
    __builtin_amdgcn_tensor_load_to_lds(g0, g1, gz, gz, 0);
#endif
}
#else
#define HAS_TDM 0
#endif

// ================= K0a: transpose+convert weights to bf16 =================
__global__ void vq_k0_weights(const float* __restrict__ W_in, const float* __restrict__ W_out,
                              bf16* __restrict__ W_inT, bf16* __restrict__ W_outT) {
    int i = blockIdx.x * blockDim.x + threadIdx.x;   // 524288 threads
    if (i < DIMq * CDq) {
        int k = i / CDq, n = i % CDq;                // W_in [DIM][CD]
        W_inT[(size_t)n * DIMq + k] = (bf16)W_in[i];
        int c = i / DIMq, d = i % DIMq;              // W_out [CD][DIM]
        W_outT[(size_t)d * CDq + c] = (bf16)W_out[i];
    }
}

// ================= K0b: embed -> bf16 + half squared norms =================
__global__ void vq_k0_embed(const float* __restrict__ embed, bf16* __restrict__ embed_b,
                            float* __restrict__ halfnorm) {
    const int r = blockIdx.x;           // 8192 rows
    const int t = threadIdx.x;          // 256 threads, 2 cols each
    const float2 v = ((const float2*)(embed + (size_t)r * CDq))[t];
    bf16* eb = embed_b + (size_t)r * CDq;
    eb[2 * t]     = (bf16)v.x;
    eb[2 * t + 1] = (bf16)v.y;
    float s = v.x * v.x + v.y * v.y;
#pragma unroll
    for (int m = 16; m >= 1; m >>= 1) s += __shfl_xor(s, m);
    __shared__ float wsum[8];
    if ((t & 31) == 0) wsum[t >> 5] = s;
    __syncthreads();
    if (t == 0) {
        float tot = 0.f;
#pragma unroll
        for (int w = 0; w < 8; ++w) tot += wsum[w];
        halfnorm[r] = 0.5f * tot;
    }
}

// ================= K1: xp = x @ W_in + b_in  (bf16 WMMA, f32 accum) =================
// grid (Mtot/128, CD/128), 256 threads. Per wave: 16 rows x 128 cols.
__global__ __launch_bounds__(256) void vq_k1_proj_in(const float* __restrict__ x,
                                                     const float* __restrict__ b_in,
                                                     const bf16* __restrict__ W_inT,
                                                     bf16* __restrict__ xp_b) {
    __shared__ __align__(16) bf16 Bs[128 * 40];   // 32 k-cols + pad, per of 128 n-rows
    const int tid = threadIdx.x, lane = tid & 31, wave = tid >> 5;
    const int half = (lane >> 4) & 1, lidx = lane & 15, kbase = half * 8;
    const int m0 = blockIdx.x * 128, n0 = blockIdx.y * 128;
    const int mrow = m0 + wave * 16 + lidx;

    v8f acc[8];
#pragma unroll
    for (int nt = 0; nt < 8; ++nt) {
        float bv = b_in[n0 + nt * 16 + lidx];
#pragma unroll
        for (int j = 0; j < 8; ++j) acc[nt][j] = bv;
    }

    for (int k = 0; k < DIMq; k += 32) {
        __syncthreads();
        for (int i = tid; i < 512; i += 256) {           // async-stage W_in^T tile [128 x 32]
            int n = i >> 2, c8 = (i & 3) * 8;
            stage_b128(&Bs[n * 40 + c8], &W_inT[(size_t)(n0 + n) * DIMq + k + c8]);
        }
        stage_wait();
        __syncthreads();

        // A fragment straight from f32 x, converted in-register (v_cvt_pk_bf16_f32)
        const float* xr = x + (size_t)mrow * DIMq + k + kbase;
        v4f a0 = *(const v4f*)xr;
        v4f a1 = *(const v4f*)(xr + 4);
        v4f a2 = *(const v4f*)(xr + 16);
        v4f a3 = *(const v4f*)(xr + 20);
        v16bf af;
#pragma unroll
        for (int i = 0; i < 4; ++i) {
            af[i] = (bf16)a0[i]; af[4 + i] = (bf16)a1[i];
            af[8 + i] = (bf16)a2[i]; af[12 + i] = (bf16)a3[i];
        }
#pragma unroll
        for (int nt = 0; nt < 8; ++nt)
            acc[nt] = wmma_bf16(af, frag16(&Bs[(nt * 16 + lidx) * 40 + kbase]), acc[nt]);
    }

    const int rbase = m0 + wave * 16 + half * 8;
#pragma unroll
    for (int nt = 0; nt < 8; ++nt) {
        int col = n0 + nt * 16 + lidx;
#pragma unroll
        for (int j = 0; j < 8; ++j)
            xp_b[(size_t)(rbase + j) * CDq + col] = (bf16)acc[nt][j];
    }
}

// ================= K2: fused score GEMM + argmax =================
// score = xp . e_k - 0.5||e_k||^2  (half-norm folded into WMMA C init).
// 128 xp rows resident in LDS; 64-code embed blocks streamed via TDM, double-buffered.
#define XS 520   // 512 + 8 elem pad -> conflict-free ds_load_b128 (TDM pad engine produces this)
#define ES 520
#define K2_EB0  (128u * XS)        // elem offset of ebuf0
#define K2_EBSZ (64u * ES)         // elems per ebuf buffer
#define K2_NB   (Kq / 64)          // 128 code blocks

__device__ __forceinline__ void k2_compute_block(const bf16* __restrict__ ebuf,
                                                 const bf16* __restrict__ arow,
                                                 const float* __restrict__ halfnorm,
                                                 int code0, int lidx, int kbase,
                                                 float* bestv, int* besti) {
    v8f acc[4];
#pragma unroll
    for (int c = 0; c < 4; ++c) {
        float hn = halfnorm[code0 + c * 16 + lidx];
#pragma unroll
        for (int j = 0; j < 8; ++j) acc[c][j] = -hn;     // fold -0.5||e||^2 into C
    }
#pragma unroll
    for (int ks = 0; ks < CDq; ks += 32) {
        v16bf af = frag16(arow + ks);                    // one A frag feeds 4 WMMAs
#pragma unroll
        for (int c = 0; c < 4; ++c)
            acc[c] = wmma_bf16(af, frag16(&ebuf[(c * 16 + lidx) * ES + kbase + ks]), acc[c]);
    }
#pragma unroll
    for (int c = 0; c < 4; ++c) {
        const int code = code0 + c * 16 + lidx;
#pragma unroll
        for (int j = 0; j < 8; ++j)
            if (acc[c][j] > bestv[j]) { bestv[j] = acc[c][j]; besti[j] = code; }
    }
}

__global__ __launch_bounds__(256) void vq_k2_argmax(const bf16* __restrict__ xp_b,
                                                    const bf16* __restrict__ embed_b,
                                                    const float* __restrict__ halfnorm,
                                                    int* __restrict__ ind_ws,
                                                    float* __restrict__ ind_out) {
    extern __shared__ __align__(16) bf16 lds[];
    bf16* xbuf = lds;                         // 128 x XS
    const int tid = threadIdx.x, lane = tid & 31, wave = tid >> 5;
    const int half = (lane >> 4) & 1, lidx = lane & 15, kbase = half * 8;
    const int r0 = blockIdx.x * 128;
    const int wl = wave * 16;
    const bf16* arow = &xbuf[(wl + lidx) * XS + kbase];

    float bestv[8]; int besti[8];
#pragma unroll
    for (int j = 0; j < 8; ++j) { bestv[j] = -3.4e38f; besti[j] = 0; }

#if HAS_TDM
    const unsigned lds0 = LDS_BYTE_OFF(lds);
    if (wave == 0) {
        tdm_load_2d(lds0, xp_b + (size_t)r0 * CDq, 128);                 // resident xp block
        tdm_load_2d(lds0 + K2_EB0 * 2, embed_b, 64);                     // code block 0
    }
    for (int cb = 0; cb < K2_NB; ++cb) {
        if (wave == 0) {
            if (cb + 1 < K2_NB) {
                tdm_load_2d(lds0 + (K2_EB0 + (unsigned)((cb + 1) & 1) * K2_EBSZ) * 2,
                            embed_b + (size_t)(cb + 1) * 64 * CDq, 64);
                __builtin_amdgcn_s_wait_tensorcnt((short)1);  // block cb landed; cb+1 in flight
            } else {
                __builtin_amdgcn_s_wait_tensorcnt((short)0);
            }
        }
        __syncthreads();                                   // publish buf[cb&1] to all waves
        k2_compute_block(&lds[K2_EB0 + (unsigned)(cb & 1) * K2_EBSZ], arow, halfnorm,
                         cb * 64, lidx, kbase, bestv, besti);
        __syncthreads();                                   // compute done before next overwrite
    }
#else
    for (int i = tid; i < 128 * 64; i += 256) {            // stage resident xp block
        int r = i >> 6, c8 = (i & 63) * 8;
        *(v8bf*)&xbuf[r * XS + c8] = *(const v8bf*)&xp_b[(size_t)(r0 + r) * CDq + c8];
    }
    for (int cb = 0; cb < K2_NB; ++cb) {
        __syncthreads();
        for (int i = tid; i < 64 * 64; i += 256) {         // stream 64-code block
            int r = i >> 6, c8 = (i & 63) * 8;
            *(v8bf*)&lds[K2_EB0 + r * ES + c8] =
                *(const v8bf*)&embed_b[((size_t)cb * 64 + r) * CDq + c8];
        }
        __syncthreads();
        k2_compute_block(&lds[K2_EB0], arow, halfnorm, cb * 64, lidx, kbase, bestv, besti);
    }
#endif

    // cross-lane argmax over 16 columns within each 16-lane half (wave32)
#pragma unroll
    for (int m = 1; m < 16; m <<= 1) {
#pragma unroll
        for (int j = 0; j < 8; ++j) {
            float ov = __shfl_xor(bestv[j], m);
            int   oi = __shfl_xor(besti[j], m);
            if (ov > bestv[j] || (ov == bestv[j] && oi < besti[j])) { bestv[j] = ov; besti[j] = oi; }
        }
    }
    if (lidx == 0) {
        int rbase = r0 + wl + half * 8;
#pragma unroll
        for (int j = 0; j < 8; ++j) {
            ind_ws[rbase + j]  = besti[j];
            ind_out[rbase + j] = (float)besti[j];
        }
    }
}

// ================= K_loss: per-row deterministic partial MSE =================
__global__ void vq_k_loss(const int* __restrict__ ind_ws, const bf16* __restrict__ embed_b,
                          const bf16* __restrict__ xp_b, float* __restrict__ part) {
    const int r = blockIdx.x, t = threadIdx.x;
    const bf16* q  = embed_b + (size_t)ind_ws[r] * CDq;
    const bf16* xr = xp_b + (size_t)r * CDq;
    float d0 = (float)q[2 * t] - (float)xr[2 * t];
    float d1 = (float)q[2 * t + 1] - (float)xr[2 * t + 1];
    float s = d0 * d0 + d1 * d1;
#pragma unroll
    for (int m = 16; m >= 1; m >>= 1) s += __shfl_xor(s, m);
    __shared__ float wsum[8];
    if ((t & 31) == 0) wsum[t >> 5] = s;
    __syncthreads();
    if (t == 0) {
        float tot = 0.f;
#pragma unroll
        for (int w = 0; w < 8; ++w) tot += wsum[w];
        part[r] = tot;
    }
}

__global__ void vq_k_loss_final(const float* __restrict__ part, float* __restrict__ loss) {
    __shared__ float sh[256];
    float s = 0.f;
    for (int r = threadIdx.x; r < Mtot; r += 256) s += part[r];   // fixed order
    sh[threadIdx.x] = s;
    __syncthreads();
    for (int st = 128; st > 0; st >>= 1) {
        if (threadIdx.x < st) sh[threadIdx.x] += sh[threadIdx.x + st];
        __syncthreads();
    }
    if (threadIdx.x == 0) *loss = sh[0] / ((float)Mtot * (float)CDq);
}

// ================= K3: out = embed[ind] @ W_out + b_out =================
// grid (Mtot/128, DIM/128), 256 threads. Gathered bf16 A rows, W_out^T via LDS.
__global__ __launch_bounds__(256) void vq_k3_proj_out(const int* __restrict__ ind_ws,
                                                      const bf16* __restrict__ embed_b,
                                                      const bf16* __restrict__ W_outT,
                                                      const float* __restrict__ b_out,
                                                      float* __restrict__ out) {
    __shared__ __align__(16) bf16 Bs[128 * 40];
    const int tid = threadIdx.x, lane = tid & 31, wave = tid >> 5;
    const int half = (lane >> 4) & 1, lidx = lane & 15, kbase = half * 8;
    const int m0 = blockIdx.x * 128, n0 = blockIdx.y * 128;
    const int mrow = m0 + wave * 16 + lidx;
    const bf16* arow = embed_b + (size_t)ind_ws[mrow] * CDq + kbase;   // gather

    v8f acc[8];
#pragma unroll
    for (int nt = 0; nt < 8; ++nt) {
        float bv = b_out[n0 + nt * 16 + lidx];
#pragma unroll
        for (int j = 0; j < 8; ++j) acc[nt][j] = bv;
    }

    for (int k = 0; k < CDq; k += 32) {
        __syncthreads();
        for (int i = tid; i < 512; i += 256) {           // async-stage W_out^T tile [128 x 32]
            int n = i >> 2, c8 = (i & 3) * 8;
            stage_b128(&Bs[n * 40 + c8], &W_outT[(size_t)(n0 + n) * CDq + k + c8]);
        }
        stage_wait();
        __syncthreads();
        v16bf af = frag16(arow + k);
#pragma unroll
        for (int nt = 0; nt < 8; ++nt)
            acc[nt] = wmma_bf16(af, frag16(&Bs[(nt * 16 + lidx) * 40 + kbase]), acc[nt]);
    }

    const int rbase = m0 + wave * 16 + half * 8;
#pragma unroll
    for (int nt = 0; nt < 8; ++nt) {
        int col = n0 + nt * 16 + lidx;
#pragma unroll
        for (int j = 0; j < 8; ++j)
            out[(size_t)(rbase + j) * DIMq + col] = acc[nt][j];
    }
}

// ================= launch =================
extern "C" void kernel_launch(void* const* d_in, const int* in_sizes, int n_in,
                              void* d_out, int out_size, void* d_ws, size_t ws_size,
                              hipStream_t stream) {
    const float* x     = (const float*)d_in[0];
    const float* W_in  = (const float*)d_in[1];
    const float* b_in  = (const float*)d_in[2];
    const float* embed = (const float*)d_in[3];
    const float* W_out = (const float*)d_in[4];
    const float* b_out = (const float*)d_in[5];

    float* out     = (float*)d_out;                       // [Mtot*DIM]
    float* ind_out = out + (size_t)Mtot * DIMq;           // [Mtot] (float-cast indices)
    float* loss    = ind_out + Mtot;                      // [1]

    char* ws = (char*)d_ws;
    bf16*  W_inT    = (bf16*)(ws + WS_WINT);
    bf16*  W_outT   = (bf16*)(ws + WS_WOUTT);
    bf16*  embed_b  = (bf16*)(ws + WS_EMBED);
    float* halfnorm = (float*)(ws + WS_HALF);
    int*   ind_ws   = (int*)(ws + WS_IND);
    bf16*  xp_b     = (bf16*)(ws + WS_XPB);
    float* part     = (float*)(ws + WS_PART);

    vq_k0_weights<<<dim3((DIMq * CDq + 255) / 256), dim3(256), 0, stream>>>(W_in, W_out, W_inT, W_outT);
    vq_k0_embed<<<dim3(Kq), dim3(256), 0, stream>>>(embed, embed_b, halfnorm);

    vq_k1_proj_in<<<dim3(Mtot / 128, CDq / 128), dim3(256), 0, stream>>>(x, b_in, W_inT, xp_b);

    size_t shbytes = (size_t)(K2_EB0 + 2 * K2_EBSZ) * sizeof(bf16);  // 266240 B (< 320KB WGP LDS)
    vq_k2_argmax<<<dim3(Mtot / 128), dim3(256), shbytes, stream>>>(xp_b, embed_b, halfnorm, ind_ws, ind_out);

    vq_k_loss<<<dim3(Mtot), dim3(256), 0, stream>>>(ind_ws, embed_b, xp_b, part);
    vq_k3_proj_out<<<dim3(Mtot / 128, DIMq / 128), dim3(256), 0, stream>>>(ind_ws, embed_b, W_outT, b_out, out);
    vq_k_loss_final<<<dim3(1), dim3(256), 0, stream>>>(part, loss);
}
